// MlpAttention_68118181314609
// MI455X (gfx1250) — matrix operand order repaired and checked by reading, not compiled, not run on hardware
//
#include <hip/hip_runtime.h>
#include <hip/hip_bf16.h>
#include <math.h>

// MlpAttention for MI455X (gfx1250, wave32, WMMA f16 16x16x32, fp32 accum).
// Pipeline:
//  0a) x f32 -> x16 (row-major f16) + x16T (per-batch transposed f16)
//  0b) Wq,Wk f32 -> WqT,WkT f16 (transposed: [out][in])
//  1)  q16 = x16 @ Wq + bq ; k16 = x16 @ Wk + bk      (WMMA)
//  2)  sig = sigmoid(q16 @ k16^T / sqrt(512))          (WMMA) -> staged in d_out attn region
//  3)  attention = softmax(sig) in place + attn16 copy
//  4)  ax16 = attention @ x  (per batch)               (WMMA)
//  5)  split-K partials of ax @ Wm                     (WMMA, streams 512MB of Wm)
//  6)  deterministic reduce of partials + bm -> out

typedef _Float16 f16;
typedef _Float16 half8 __attribute__((ext_vector_type(8)));
typedef _Float16 v16h  __attribute__((ext_vector_type(16)));
typedef float    v8f   __attribute__((ext_vector_type(8)));
typedef float    f32x4 __attribute__((ext_vector_type(4)));

#define B_  32
#define N_  512
#define H_  512
#define KNH 262144          // N_*H_
#define NCHUNK 128          // split-K chunks for final GEMM
#define KC  2048            // K per chunk (KC*NCHUNK == KNH)

// ---- WMMA fragment loaders (contiguous k order: lane half h holds k 16h..16h+15, same for A and B) ----

// A (16x32, MxK), row-major source with leading dim lda.
// lane<16: halves[0..15]=K0..15 (M=lane)
// lane>=16: halves[0..15]=K16..31 (M=lane-16)
__device__ inline v16h load_a_frag(const f16* __restrict__ A, int lda, int row0,
                                   int k0, int lane) {
    int m = row0 + (lane & 15);
    int koff = (lane < 16) ? 0 : 16;
    const f16* p = A + (size_t)m * lda + k0 + koff;
    union { v16h v; half8 h[2]; } u;
    u.h[0] = *(const half8*)(p);
    u.h[1] = *(const half8*)(p + 8);
    return u.v;
}

// B (32x16, KxN) loaded from BT = B^T stored row-major [N][K] with ldb.
// lane<16: halves[0..15] = K0..15 of column n=lane (contiguous 32B)
// lane>=16: halves[0..15] = K16..31 of column n=lane-16
__device__ inline v16h load_bt_frag(const f16* __restrict__ BT, int ldb, int col0,
                                    int k0, int lane) {
    int n = col0 + (lane & 15);
    int kbase = (lane < 16) ? 0 : 16;
    return *(const v16h*)(BT + (size_t)n * ldb + k0 + kbase);
}

// ---------------- 0a: convert + transpose x ----------------
__global__ void cvt_x_kernel(const float* __restrict__ x,
                             f16* __restrict__ x16, f16* __restrict__ x16T) {
    __shared__ f16 tile[32][33];
    int b  = blockIdx.z;
    int j0 = blockIdx.y * 32;   // node
    int h0 = blockIdx.x * 32;   // feature
    int tx = threadIdx.x, ty = threadIdx.y;
    const float* xb = x + (size_t)b * N_ * H_;
    f16* x16b = x16 + (size_t)b * N_ * H_;
    f16* xtb  = x16T + (size_t)b * N_ * H_;
#pragma unroll
    for (int r = 0; r < 4; ++r) {
        int j = j0 + ty + r * 8;
        f16 v = (f16)xb[(size_t)j * H_ + h0 + tx];
        x16b[(size_t)j * H_ + h0 + tx] = v;
        tile[ty + r * 8][tx] = v;
    }
    __syncthreads();
#pragma unroll
    for (int r = 0; r < 4; ++r) {
        int h = h0 + ty + r * 8;
        xtb[(size_t)h * N_ + j0 + tx] = tile[tx][ty + r * 8];
    }
}

// ---------------- 0b: convert + transpose Wq/Wk ----------------
__global__ void cvt_w_kernel(const float* __restrict__ Wq, const float* __restrict__ Wk,
                             f16* __restrict__ WqT, f16* __restrict__ WkT) {
    __shared__ f16 tile[32][33];
    const float* W = blockIdx.z ? Wk : Wq;
    f16* WT = blockIdx.z ? WkT : WqT;
    int k0 = blockIdx.y * 32;   // input dim (rows of W)
    int n0 = blockIdx.x * 32;   // output dim (cols of W)
    int tx = threadIdx.x, ty = threadIdx.y;
#pragma unroll
    for (int r = 0; r < 4; ++r) {
        int k = k0 + ty + r * 8;
        tile[ty + r * 8][tx] = (f16)W[(size_t)k * H_ + n0 + tx];
    }
    __syncthreads();
#pragma unroll
    for (int r = 0; r < 4; ++r) {
        int n = n0 + ty + r * 8;
        WT[(size_t)n * H_ + k0 + tx] = tile[tx][ty + r * 8];
    }
}

// ---------------- 1: q/k projection (WMMA) ----------------
// grid(1024 mtiles, 4 ngroups, 2 which) x 256 threads (8 waves, 16 cols each)
__global__ void proj_kernel(const f16* __restrict__ x16,
                            const f16* __restrict__ WqT, const f16* __restrict__ WkT,
                            const float* __restrict__ bq, const float* __restrict__ bk,
                            f16* __restrict__ q16, f16* __restrict__ k16) {
    __shared__ __align__(16) f16 sTile[16][136];
    int mtile  = blockIdx.x;
    int ngroup = blockIdx.y;
    int which  = blockIdx.z;
    const f16* WT = which ? WkT : WqT;
    const float* bias = which ? bk : bq;
    f16* out = which ? k16 : q16;
    int t = threadIdx.x, lane = t & 31, wave = t >> 5;
    int row0 = mtile * 16;
    int col0 = ngroup * 128 + wave * 16;
    v8f acc = {};
    for (int k0 = 0; k0 < H_; k0 += 32) {
        v16h a = load_a_frag(x16, H_, row0, k0, lane);
        v16h b = load_bt_frag(WT, H_, col0, k0, lane);
        acc = __builtin_amdgcn_wmma_f32_16x16x32_f16(false, a, false, b,
                                                     (short)0, acc, false, false);
    }
    float bv = bias[col0 + (lane & 15)];
    int nLoc = wave * 16 + (lane & 15);
    int mhi = (lane >> 4) << 3;
#pragma unroll
    for (int r = 0; r < 8; ++r)
        sTile[mhi + r][nLoc] = (f16)(acc[r] + bv);
    __syncthreads();
    int row = t >> 4, seg = t & 15;   // 16 rows x 16 segs of 8 halves
    half8 v = *(const half8*)&sTile[row][seg * 8];
    *(half8*)(out + (size_t)(row0 + row) * H_ + ngroup * 128 + seg * 8) = v;
}

// ---------------- 2: sigmoid(q k^T / sqrt(d)) (WMMA) ----------------
// grid(32 itiles, 4 jgroups, 32 batches) x 256 threads
__global__ void scores_kernel(const f16* __restrict__ q16, const f16* __restrict__ k16,
                              float* __restrict__ sig_out) {
    __shared__ __align__(16) float sTile[16][132];
    int itile = blockIdx.x, jgroup = blockIdx.y, b = blockIdx.z;
    int t = threadIdx.x, lane = t & 31, wave = t >> 5;
    const f16* qb = q16 + (size_t)b * N_ * H_;
    const f16* kb = k16 + (size_t)b * N_ * H_;   // k is already B^T for q@k^T
    int row0 = itile * 16;
    int col0 = jgroup * 128 + wave * 16;
    v8f acc = {};
    for (int k0 = 0; k0 < H_; k0 += 32) {
        v16h a = load_a_frag(qb, H_, row0, k0, lane);
        v16h bfr = load_bt_frag(kb, H_, col0, k0, lane);
        acc = __builtin_amdgcn_wmma_f32_16x16x32_f16(false, a, false, bfr,
                                                     (short)0, acc, false, false);
    }
    const float scale = 0.044194173824159216f;   // 1/sqrt(512)
    int nLoc = wave * 16 + (lane & 15);
    int mhi = (lane >> 4) << 3;
#pragma unroll
    for (int r = 0; r < 8; ++r) {
        float s = acc[r] * scale;
        sTile[mhi + r][nLoc] = 1.0f / (1.0f + __expf(-s));
    }
    __syncthreads();
    int row = t >> 4, seg = t & 15;
    const f32x4* src = (const f32x4*)&sTile[row][seg * 8];
    float* dst = sig_out + ((size_t)(b * N_ + row0 + row)) * N_ + jgroup * 128 + seg * 8;
    ((f32x4*)dst)[0] = src[0];
    ((f32x4*)dst)[1] = src[1];
}

// ---------------- 3: row softmax in place + f16 copy ----------------
// one block (256 threads) per row of 512
__global__ void softmax_kernel(float* __restrict__ attn, f16* __restrict__ attn16) {
    __shared__ float red[8];
    int row = blockIdx.x;
    int t = threadIdx.x;
    float* p = attn + (size_t)row * N_;
    float v0 = p[t], v1 = p[t + 256];
    float m = fmaxf(v0, v1);
#pragma unroll
    for (int off = 16; off > 0; off >>= 1) m = fmaxf(m, __shfl_xor(m, off, 32));
    if ((t & 31) == 0) red[t >> 5] = m;
    __syncthreads();
    if (t < 8) {
        float mm = red[t];
#pragma unroll
        for (int off = 4; off > 0; off >>= 1) mm = fmaxf(mm, __shfl_xor(mm, off, 32));
        if (t == 0) red[0] = mm;
    }
    __syncthreads();
    m = red[0];
    __syncthreads();
    float e0 = __expf(v0 - m), e1 = __expf(v1 - m);
    float s = e0 + e1;
#pragma unroll
    for (int off = 16; off > 0; off >>= 1) s += __shfl_xor(s, off, 32);
    if ((t & 31) == 0) red[t >> 5] = s;
    __syncthreads();
    if (t < 8) {
        float ss = red[t];
#pragma unroll
        for (int off = 4; off > 0; off >>= 1) ss += __shfl_xor(ss, off, 32);
        if (t == 0) red[0] = ss;
    }
    __syncthreads();
    float inv = 1.0f / red[0];
    float a0 = e0 * inv, a1 = e1 * inv;
    p[t] = a0; p[t + 256] = a1;
    f16* p16 = attn16 + (size_t)row * N_;
    p16[t] = (f16)a0; p16[t + 256] = (f16)a1;
}

// ---------------- 4: attention @ x (WMMA, per batch) ----------------
__global__ void attnx_kernel(const f16* __restrict__ attn16, const f16* __restrict__ x16T,
                             f16* __restrict__ ax16) {
    __shared__ __align__(16) f16 sTile[16][136];
    int itile = blockIdx.x, hgroup = blockIdx.y, b = blockIdx.z;
    int t = threadIdx.x, lane = t & 31, wave = t >> 5;
    const f16* ab  = attn16 + (size_t)b * N_ * N_;
    const f16* xtb = x16T + (size_t)b * N_ * H_;   // [h][j] = B^T
    int row0 = itile * 16;
    int col0 = hgroup * 128 + wave * 16;
    v8f acc = {};
    for (int k0 = 0; k0 < N_; k0 += 32) {
        v16h a = load_a_frag(ab, N_, row0, k0, lane);
        v16h bfr = load_bt_frag(xtb, N_, col0, k0, lane);
        acc = __builtin_amdgcn_wmma_f32_16x16x32_f16(false, a, false, bfr,
                                                     (short)0, acc, false, false);
    }
    int nLoc = wave * 16 + (lane & 15);
    int mhi = (lane >> 4) << 3;
#pragma unroll
    for (int r = 0; r < 8; ++r)
        sTile[mhi + r][nLoc] = (f16)acc[r];
    __syncthreads();
    int row = t >> 4, seg = t & 15;
    half8 v = *(const half8*)&sTile[row][seg * 8];
    *(half8*)(ax16 + ((size_t)(b * N_ + row0 + row)) * H_ + hgroup * 128 + seg * 8) = v;
}

// ---------------- 5: final GEMM split-K partials (WMMA, streams Wm) ----------------
// out_part[chunk][m(32)][c(512)]; grid(NCHUNK, 2 mtiles, 4 ngroups) x 256 threads
__global__ void final_partial_kernel(const f16* __restrict__ ax16,
                                     const float* __restrict__ Wm,
                                     float* __restrict__ part) {
    int chunk = blockIdx.x, mtile = blockIdx.y, ngroup = blockIdx.z;
    int t = threadIdx.x, lane = t & 31, wave = t >> 5;
    int col = ngroup * 128 + wave * 16 + (lane & 15);
    int row0 = mtile * 16;
    int kstart = chunk * KC;
    int kbase = (lane < 16) ? 0 : 16;
    v8f acc = {};
    for (int ks = 0; ks < KC; ks += 32) {
        int k0 = kstart + ks;
        v16h a = load_a_frag(ax16, KNH, row0, k0, lane);
        v16h bfr;
        const float* wp = Wm + (size_t)(k0 + kbase) * H_ + col;
#pragma unroll
        for (int i = 0; i < 16; ++i) bfr[i] = (f16)wp[(size_t)i * H_];
        acc = __builtin_amdgcn_wmma_f32_16x16x32_f16(false, a, false, bfr,
                                                     (short)0, acc, false, false);
    }
    int mhi = (lane >> 4) << 3;
    float* pp = part + (size_t)chunk * (32 * 512);
#pragma unroll
    for (int r = 0; r < 8; ++r) {
        int m = row0 + mhi + r;
        pp[(size_t)m * H_ + col] = acc[r];
    }
}

// ---------------- 6: reduce partials + bias ----------------
__global__ void final_reduce_kernel(const float* __restrict__ part,
                                    const float* __restrict__ bm,
                                    float* __restrict__ out) {
    int idx = blockIdx.x * 256 + threadIdx.x;   // 0..16383 over [32][512]
    int c = idx & 511;
    float s = bm[c];
    for (int ch = 0; ch < NCHUNK; ++ch)
        s += part[(size_t)ch * 16384 + idx];
    out[idx] = s;
}

// ---------------- launch ----------------
extern "C" void kernel_launch(void* const* d_in, const int* in_sizes, int n_in,
                              void* d_out, int out_size, void* d_ws, size_t ws_size,
                              hipStream_t stream) {
    const float* x  = (const float*)d_in[0];
    const float* Wq = (const float*)d_in[1];
    const float* bq = (const float*)d_in[2];
    const float* Wk = (const float*)d_in[3];
    const float* bk = (const float*)d_in[4];
    const float* Wm = (const float*)d_in[5];
    const float* bm = (const float*)d_in[6];
    float* out  = (float*)d_out;            // [32,512]
    float* attn = out + 32 * 512;           // [32,512,512]

    char* ws = (char*)d_ws;
    const size_t SZ_BNH16 = (size_t)B_ * N_ * H_ * sizeof(f16);   // 16 MB
    const size_t SZ_W16   = (size_t)H_ * H_ * sizeof(f16);        // 512 KB
    size_t o = 0;
    f16* x16    = (f16*)(ws + o); o += SZ_BNH16;
    f16* x16T   = (f16*)(ws + o); o += SZ_BNH16;
    f16* WqT    = (f16*)(ws + o); o += SZ_W16;
    f16* WkT    = (f16*)(ws + o); o += SZ_W16;
    f16* q16    = (f16*)(ws + o); o += SZ_BNH16;
    f16* k16    = (f16*)(ws + o); o += SZ_BNH16;
    f16* attn16 = (f16*)(ws + o); o += SZ_BNH16;   // B*N*N == B*N*H here
    f16* ax16   = (f16*)(ws + o); o += SZ_BNH16;
    float* part = (float*)(ws + o); o += (size_t)NCHUNK * 32 * 512 * sizeof(float);
    (void)ws_size; (void)in_sizes; (void)n_in; (void)out_size;

    cvt_x_kernel<<<dim3(H_ / 32, N_ / 32, B_), dim3(32, 8), 0, stream>>>(x, x16, x16T);
    cvt_w_kernel<<<dim3(H_ / 32, H_ / 32, 2), dim3(32, 8), 0, stream>>>(Wq, Wk, WqT, WkT);
    proj_kernel<<<dim3((B_ * N_) / 16, 4, 2), dim3(256), 0, stream>>>(
        x16, WqT, WkT, bq, bk, q16, k16);
    scores_kernel<<<dim3(N_ / 16, 4, B_), dim3(256), 0, stream>>>(q16, k16, attn);
    softmax_kernel<<<dim3(B_ * N_), dim3(256), 0, stream>>>(attn, attn16);
    attnx_kernel<<<dim3(N_ / 16, 4, B_), dim3(256), 0, stream>>>(attn16, x16T, ax16);
    final_partial_kernel<<<dim3(NCHUNK, 2, 4), dim3(256), 0, stream>>>(ax16, Wm, part);
    final_reduce_kernel<<<dim3(64), dim3(256), 0, stream>>>(part, bm, out);
}
